// Criterion_24472723653292
// MI455X (gfx1250) — compile-verified, matrix operand order checked by reading, not run on hardware
//
#include <hip/hip_runtime.h>
#include <stdint.h>

// ---------------------------------------------------------------------------
// Types for the CDNA5 WMMA builtin (gfx1250, wave32):
//   v_wmma_f32_16x16x32_bf16:  A 16x32 bf16, B 32x16 bf16, C/D 16x16 f32
// ---------------------------------------------------------------------------
typedef __bf16 bf16t;
typedef __attribute__((ext_vector_type(8)))  __bf16 v8bf;
typedef __attribute__((ext_vector_type(16))) __bf16 v16bf;
typedef __attribute__((ext_vector_type(8)))  float  v8f;

__device__ __forceinline__ unsigned short f32_to_bf16_rne(float f) {
    unsigned u = __float_as_uint(f);
    u += 0x7fffu + ((u >> 16) & 1u);          // round-to-nearest-even
    return (unsigned short)(u >> 16);
}
__device__ __forceinline__ float bf16_to_f32(unsigned short h) {
    return __uint_as_float(((unsigned)h) << 16);
}

// Load one 16x32 bf16 WMMA fragment for this lane: two contiguous 8-elem
// (16B) groups per the ISA 16-bit A/B layout. Compiles to 2x global_load_b128
// with immediate offsets when o0/o1 are compile-time constants.
__device__ __forceinline__ v16bf load_frag(const bf16t* __restrict__ rowbase,
                                           int o0, int o1) {
    v8bf p0 = *(const v8bf*)(rowbase + o0);
    v8bf p1 = *(const v8bf*)(rowbase + o1);
    v16bf f;
    #pragma unroll
    for (int e = 0; e < 8; ++e) { f[e] = p0[e]; f[e + 8] = p1[e]; }
    return f;
}

// ---------------------------------------------------------------------------
// Kernel 0: init argmin buffer (pack(+inf, 0xFFFFFFFF)) and class histogram
// ---------------------------------------------------------------------------
__global__ void k_init(unsigned long long* __restrict__ best,
                       int* __restrict__ cls, int B, int C) {
    int i = blockIdx.x * blockDim.x + threadIdx.x;
    if (i < B) best[i] = 0x7F800000FFFFFFFFull;
    if (i < C) cls[i]  = 0;
}

// ---------------------------------------------------------------------------
// Kernel 1: per-row squared norm (exact fp32), bf16 hi/lo split, class counts
// One wave per row.
// ---------------------------------------------------------------------------
__global__ void k_prep(const float* __restrict__ x, const int* __restrict__ labels,
                       unsigned short* __restrict__ xhi, unsigned short* __restrict__ xlo,
                       float* __restrict__ sq, int* __restrict__ cls, int B, int D) {
    int row  = (blockIdx.x * blockDim.x + threadIdx.x) >> 5;
    int lane = threadIdx.x & 31;
    if (row >= B) return;
    const float* r = x + (size_t)row * D;
    float s = 0.f;
    for (int k = lane; k < D; k += 32) {
        float v = r[k];
        unsigned short h = f32_to_bf16_rne(v);
        float res = v - bf16_to_f32(h);
        unsigned short l = f32_to_bf16_rne(res);
        xhi[(size_t)row * D + k] = h;
        xlo[(size_t)row * D + k] = l;
        s += v * v;
    }
    for (int m = 16; m >= 1; m >>= 1) s += __shfl_xor(s, m, 32);
    if (lane == 0) {
        sq[row] = s;
        atomicAdd(&cls[labels[row]], 1);
    }
}

// ---------------------------------------------------------------------------
// Kernel 2: fused Gram-GEMM + masked argmin (hardest-positive mining).
// Block = 256 threads = 8 waves, block tile 128(i) x 64(j).
// Each wave owns a 32x32 output region = 2x2 WMMA subtiles (4 accumulators).
// Split-bf16 for ~fp32 accuracy: G = Ah*Bh + Ah*Bl + Al*Bh.
// DIMC > 0: compile-time K extent -> full unroll, immediate-offset b128 loads,
// no per-chunk address VALU and no WMMA->VALU WAR hazard NOPs.
// ---------------------------------------------------------------------------
template <int DIMC>
__global__ __launch_bounds__(256) void k_mine(
        const unsigned short* __restrict__ xhi_, const unsigned short* __restrict__ xlo_,
        const float* __restrict__ sq, const int* __restrict__ labels,
        unsigned long long* __restrict__ best, int B, int D) {
    __shared__ int   sLabI[128];
    __shared__ int   sLabJ[64];
    __shared__ float sSqI[128];
    __shared__ float sSqJ[64];

    const int Dloc = (DIMC > 0) ? DIMC : D;

    const int i0 = blockIdx.x * 128;
    const int j0 = blockIdx.y * 64;
    const int t  = threadIdx.x;
    if (t < 128)       { sLabI[t] = labels[i0 + t]; sSqI[t] = sq[i0 + t]; }
    else if (t < 192)  { int u = t - 128; sLabJ[u] = labels[j0 + u]; sSqJ[u] = sq[j0 + u]; }
    __syncthreads();

    const int wv   = t >> 5;         // wave 0..7
    const int lane = t & 31;
    const int wm   = wv & 3;         // 4 wave-tiles along i (32 rows each)
    const int wn   = wv >> 2;        // 2 wave-tiles along j (32 cols each)
    const int h    = lane >> 4;      // K-half select (ISA A/B layout)
    const int r    = lane & 15;      // M (for A) / N (for B)

    // Per-lane row pointers for the 2 A-subtiles and 2 B-subtiles (hi & lo).
    // Fold the lane's K-half offset (8*h) into the base so the unrolled loop
    // uses pure immediate offsets.
    const bf16t* ARhi[2]; const bf16t* ARlo[2];
    const bf16t* BRhi[2]; const bf16t* BRlo[2];
    #pragma unroll
    for (int s = 0; s < 2; ++s) {
        const int irow = i0 + (wm * 2 + s) * 16 + r;
        const int jrow = j0 + (wn * 2 + s) * 16 + r;
        ARhi[s] = (const bf16t*)xhi_ + (size_t)irow * Dloc + 8 * h;
        ARlo[s] = (const bf16t*)xlo_ + (size_t)irow * Dloc + 8 * h;
        BRhi[s] = (const bf16t*)xhi_ + (size_t)jrow * Dloc + 8 * h;
        BRlo[s] = (const bf16t*)xlo_ + (size_t)jrow * Dloc + 8 * h;
    }

    v8f acc[2][2] = {};
    #pragma unroll
    for (int kc = 0; kc < Dloc; kc += 32) {
        v16bf Ah[2], Al[2], Bh[2], Bl[2];
        #pragma unroll
        for (int s = 0; s < 2; ++s) {
            Ah[s] = load_frag(ARhi[s], kc, kc + 16);
            Al[s] = load_frag(ARlo[s], kc, kc + 16);
            Bh[s] = load_frag(BRhi[s], kc, kc + 16);
            Bl[s] = load_frag(BRlo[s], kc, kc + 16);
        }
        // 12 WMMAs per chunk vs 16 b128 loads
        #pragma unroll
        for (int sm = 0; sm < 2; ++sm) {
            #pragma unroll
            for (int sn = 0; sn < 2; ++sn) {
                acc[sm][sn] = __builtin_amdgcn_wmma_f32_16x16x32_bf16(
                    false, Ah[sm], false, Bh[sn], (short)0, acc[sm][sn], false, false);
                acc[sm][sn] = __builtin_amdgcn_wmma_f32_16x16x32_bf16(
                    false, Ah[sm], false, Bl[sn], (short)0, acc[sm][sn], false, false);
                acc[sm][sn] = __builtin_amdgcn_wmma_f32_16x16x32_bf16(
                    false, Al[sm], false, Bh[sn], (short)0, acc[sm][sn], false, false);
            }
        }
    }

    // C/D layout: lane n = lane%16, VGPR v -> m = v + 8*(lane/16)
    #pragma unroll
    for (int sn = 0; sn < 2; ++sn) {
        const int nloc  = (wn * 2 + sn) * 16 + r;
        const int j     = j0 + nloc;
        const int labj  = sLabJ[nloc];
        const float sqj = sSqJ[nloc];
        #pragma unroll
        for (int sm = 0; sm < 2; ++sm) {
            #pragma unroll
            for (int v = 0; v < 8; ++v) {
                const int mloc = (wm * 2 + sm) * 16 + v + 8 * h;
                const int i    = i0 + mloc;
                if (sLabI[mloc] == labj && i != j) {
                    float d2 = fmaxf(sSqI[mloc] + sqj - 2.0f * acc[sm][sn][v], 0.0f);
                    unsigned long long pk =
                        ((unsigned long long)__float_as_uint(d2) << 32) | (unsigned)j;
                    atomicMin(&best[i], pk);   // order-independent -> deterministic
                }
            }
        }
    }
}

// ---------------------------------------------------------------------------
// Kernel 3: per-triplet exact fp32 distances + softplus. One wave per triplet.
// Writes per-triplet loss/valid so the final sum is order-deterministic.
// ---------------------------------------------------------------------------
__global__ void k_loss(const float* __restrict__ x, const int* __restrict__ labels,
                       const int* __restrict__ anchors, const int* __restrict__ negatives,
                       const unsigned long long* __restrict__ best,
                       const int* __restrict__ cls,
                       float* __restrict__ perArr, float* __restrict__ cntArr,
                       int B, int D) {
    int tIdx = (blockIdx.x * blockDim.x + threadIdx.x) >> 5;
    int lane = threadIdx.x & 31;
    if (tIdx >= B) return;
    int a = anchors[tIdx];
    unsigned pj = (unsigned)(best[tIdx] & 0xffffffffu);
    int p  = (pj < (unsigned)B) ? (int)pj : 0;   // matches argmin-of-all-inf -> 0
    int ng = negatives[tIdx];

    const float* xa = x + (size_t)a  * D;
    const float* xp = x + (size_t)p  * D;
    const float* xn = x + (size_t)ng * D;
    float sp = 0.f, sn = 0.f;
    for (int k = lane; k < D; k += 32) {
        float va = xa[k];
        float dp = va - xp[k]; sp += dp * dp;
        float dn = va - xn[k]; sn += dn * dn;
    }
    for (int m = 16; m >= 1; m >>= 1) {
        sp += __shfl_xor(sp, m, 32);
        sn += __shfl_xor(sn, m, 32);
    }
    if (lane == 0) {
        bool valid = (cls[labels[tIdx]] - 1) > 1;
        float dap = sqrtf(fmaxf(sp, 1e-12f));
        float dan = sqrtf(fmaxf(sn, 1e-12f));
        // (s_an - s_ap)/TEMP = ((dap - dan)/2)/0.05 = (dap - dan)*10
        float z   = (dap - dan) * 10.0f;
        float per = fmaxf(z, 0.f) + log1pf(expf(-fabsf(z)));   // stable softplus
        perArr[tIdx] = valid ? per  : 0.0f;
        cntArr[tIdx] = valid ? 1.0f : 0.0f;
    }
}

// ---------------------------------------------------------------------------
// Kernel 4: deterministic fixed-order reduction -> loss scalar
// ---------------------------------------------------------------------------
__global__ void k_final(const float* __restrict__ perArr, const float* __restrict__ cntArr,
                        float* __restrict__ out, int B) {
    __shared__ float sp[256];
    __shared__ float sc[256];
    int t = threadIdx.x;
    float a = 0.f, c = 0.f;
    for (int i = t; i < B; i += 256) { a += perArr[i]; c += cntArr[i]; }
    sp[t] = a; sc[t] = c;
    __syncthreads();
    for (int s = 128; s > 0; s >>= 1) {
        if (t < s) { sp[t] += sp[t + s]; sc[t] += sc[t + s]; }
        __syncthreads();
    }
    if (t == 0) out[0] = sp[0] / sc[0];
}

// ---------------------------------------------------------------------------
extern "C" void kernel_launch(void* const* d_in, const int* in_sizes, int n_in,
                              void* d_out, int out_size, void* d_ws, size_t ws_size,
                              hipStream_t stream) {
    const float* batch     = (const float*)d_in[0];
    const int*   labels    = (const int*)d_in[1];
    const int*   anchors   = (const int*)d_in[2];
    const int*   negatives = (const int*)d_in[3];

    const int B = in_sizes[1];            // 4096
    const int D = in_sizes[0] / B;        // 512
    const int C = 256;                    // histogram upper bound (>=128 classes)

    // -------- carve workspace (aligned to 256B) --------
    char* ws = (char*)d_ws;
    size_t off = 0;
    auto alloc = [&](size_t bytes) -> void* {
        void* p = ws + off;
        off += (bytes + 255) & ~(size_t)255;
        return p;
    };
    unsigned short*     xhi  = (unsigned short*)alloc((size_t)B * D * 2);
    unsigned short*     xlo  = (unsigned short*)alloc((size_t)B * D * 2);
    float*              sq   = (float*)alloc((size_t)B * 4);
    unsigned long long* best = (unsigned long long*)alloc((size_t)B * 8);
    int*                cls  = (int*)alloc((size_t)C * 4);
    float*              perA = (float*)alloc((size_t)B * 4);
    float*              cntA = (float*)alloc((size_t)B * 4);
    (void)ws_size; (void)n_in; (void)out_size;

    // -------- pipeline --------
    k_init<<<(B + 255) / 256, 256, 0, stream>>>(best, cls, B, C);
    k_prep<<<B / 8, 256, 0, stream>>>(batch, labels, xhi, xlo, sq, cls, B, D);

    dim3 g2(B / 128, B / 64);             // 32 x 64 blocks, 8 waves each
    if (D == 512) {
        k_mine<512><<<g2, 256, 0, stream>>>(xhi, xlo, sq, labels, best, B, D);
    } else {
        k_mine<0><<<g2, 256, 0, stream>>>(xhi, xlo, sq, labels, best, B, D);
    }

    k_loss<<<B / 8, 256, 0, stream>>>(batch, labels, anchors, negatives, best, cls,
                                      perA, cntA, B, D);
    k_final<<<1, 256, 0, stream>>>(perA, cntA, (float*)d_out, B);
}
